// VQVAE_52450140618792
// MI455X (gfx1250) — matrix operand order repaired and checked by reading, not compiled, not run on hardware
//
#include <hip/hip_runtime.h>

// VQ-VAE codebook quantization for MI455X (gfx1250, wave32, WMMA + async-LDS).
//   encoded:  [8192,6,6,128] f32 -> N = 294912 rows, D = 128
//   codebook: [512,128] f32      -> K = 512
// out = codebook[argmin_k(||c_k||^2 - 2 x.c_k)]   (||x||^2 is row-constant)
//
// Roofline: 38.7 GFLOP over ~302 MB HBM -> 13 us memory floor, AI ~128 FLOP/B.
// f16 WMMA with hi/lo split (3 products) keeps ~22 mantissa bits on the cross
// term (argmin-safe) at ~2.6x the f32-WMMA (16x16x4 K-depth) rate.
// Codebook hi/lo f16 is precomputed once into d_ws; blocks stage it into LDS
// with global_load_async_to_lds_b128 (ASYNCcnt), double-buffered to overlap
// the L2 stream with the WMMA chain (only 1 WG/WGP at 192 KB LDS).

typedef _Float16 v16h __attribute__((ext_vector_type(16)));
typedef float    v8f  __attribute__((ext_vector_type(8)));

#define D      128
#define KCODES 512
#define MT     128          // rows per workgroup (8 waves x 16)
#define KT     128          // codes per LDS chunk
#define NCHUNK (KCODES / KT)

// ---------------- CDNA5 async helpers (inline asm per ISA 15.18.3) ----------
__device__ __forceinline__ void async_ldg_b128(const void* gaddr, unsigned ldsByteAddr) {
    // LDS[vdst] = MEM[vaddr], tracked by ASYNCcnt
    asm volatile("global_load_async_to_lds_b128 %0, %1, off"
                 :: "v"(ldsByteAddr), "v"(gaddr) : "memory");
}
template <int N>
__device__ __forceinline__ void wait_asynccnt() {
    asm volatile("s_wait_asynccnt %0" :: "n"(N) : "memory");
}

// ---------------- prep: f16 hi/lo codebook + norms into workspace -----------
__global__ void vq_prep_kernel(const float* __restrict__ cb,
                               _Float16* __restrict__ cbHi,
                               _Float16* __restrict__ cbLo,
                               float* __restrict__ cnorm) {
    int k = blockIdx.x;          // one code per block
    int t = threadIdx.x;         // 128 threads = 4 waves
    float v = cb[k * D + t];
    _Float16 hi = (_Float16)v;
    _Float16 lo = (_Float16)(v - (float)hi);
    cbHi[k * D + t] = hi;
    cbLo[k * D + t] = lo;

    float s = v * v;
#pragma unroll
    for (int m = 16; m >= 1; m >>= 1) s += __shfl_xor(s, m, 32);
    __shared__ float part[4];
    if ((t & 31) == 0) part[t >> 5] = s;
    __syncthreads();
    if (t == 0) cnorm[k] = part[0] + part[1] + part[2] + part[3];
}

// ---------------- main kernel ----------------
__launch_bounds__(256, 1)
__global__ void vq_main_kernel(const float* __restrict__ enc,
                               const float* __restrict__ cb,
                               const _Float16* __restrict__ cbHi,
                               const _Float16* __restrict__ cbLo,
                               const float* __restrict__ cnorm,
                               float* __restrict__ out) {
    extern __shared__ char smem[];
    _Float16* Ahi  = (_Float16*)smem;            // MT*D halves (32 KB)
    _Float16* Alo  = Ahi + MT * D;               // 32 KB
    _Float16* Bbuf = Alo + MT * D;               // 2 bufs x (hi+lo) x KT*D (128 KB)

    const int tid  = threadIdx.x;
    const int lane = tid & 31;
    const int wave = tid >> 5;      // 0..7
    const int lr   = lane & 15;     // A: M row | B,C/D: N column
    const int g    = lane >> 4;     // half-wave select

    const long rowBase = (long)blockIdx.x * MT;

    // ---- Kick off async stage of B chunk 0 into buffer 0 (16 b128 / thread) ----
    {
        const char* gh = (const char*)cbHi;      // chunk 0 base
        const char* gl = (const char*)cbLo;
        unsigned lh = (unsigned)(size_t)(Bbuf);              // buf0 hi
        unsigned ll = (unsigned)(size_t)(Bbuf + KT * D);     // buf0 lo
#pragma unroll
        for (int j = 0; j < 8; ++j) {
            int off = (tid + 256 * j) * 16;
            async_ldg_b128(gh + off, lh + off);
            async_ldg_b128(gl + off, ll + off);
        }
    }

    // ---- Stage A rows -> LDS, f32 -> f16 hi + residual lo (coalesced) ----
    for (int i = tid; i < MT * D; i += 256) {
        float v = enc[rowBase * D + i];
        _Float16 hi = (_Float16)v;
        _Float16 lo = (_Float16)(v - (float)hi);
        Ahi[i] = hi; Alo[i] = lo;
    }
    __syncthreads();

    // ---- Build A fragments in registers: 4 K-steps x {hi,lo} ----
    // 16-bit A 16x32, contiguous k order used here (B matches): M = lane&15; half h -> K = 32*s + 16*(lane>>4) + h
    v16h aHi[4], aLo[4];
    {
        const unsigned* A32h = (const unsigned*)Ahi;
        const unsigned* A32l = (const unsigned*)Alo;
        const int mbase = (wave * 16 + lr) * D;
#pragma unroll
        for (int s = 0; s < 4; ++s) {
            unsigned* ah = (unsigned*)&aHi[s];
            unsigned* al = (unsigned*)&aLo[s];
#pragma unroll
            for (int p = 0; p < 8; ++p) {
                int h  = 2 * p;
                int Kk = 32 * s + g * 16 + h;
                ah[p] = A32h[(mbase + Kk) >> 1];   // pair (K, K+1), K even
                al[p] = A32l[(mbase + Kk) >> 1];
            }
        }
    }

    float minv[8];
    int   mini[8];
#pragma unroll
    for (int r = 0; r < 8; ++r) { minv[r] = 3.4e38f; mini[r] = 0; }

    for (int chunk = 0; chunk < NCHUNK; ++chunk) {
        // ---- Prefetch next chunk into the other buffer (async, no wait) ----
        if (chunk + 1 < NCHUNK) {
            const char* gh = (const char*)(cbHi + (chunk + 1) * KT * D);
            const char* gl = (const char*)(cbLo + (chunk + 1) * KT * D);
            _Float16* nb = Bbuf + ((chunk + 1) & 1) * (2 * KT * D);
            unsigned lh = (unsigned)(size_t)nb;
            unsigned ll = (unsigned)(size_t)(nb + KT * D);
#pragma unroll
            for (int j = 0; j < 8; ++j) {
                int off = (tid + 256 * j) * 16;
                async_ldg_b128(gh + off, lh + off);
                async_ldg_b128(gl + off, ll + off);
            }
            wait_asynccnt<16>();    // current chunk's 16 loads retired; prefetch in flight
        } else {
            wait_asynccnt<0>();
        }
        __syncthreads();            // publish current chunk's LDS to all waves

        const _Float16* BhiL = Bbuf + (chunk & 1) * (2 * KT * D);
        const _Float16* BloL = BhiL + KT * D;
        const uint4* Bh4 = (const uint4*)BhiL;
        const uint4* Bl4 = (const uint4*)BloL;

#pragma unroll
        for (int ct = 0; ct < KT / 16; ++ct) {
            // two accumulators to double ILP on the XDL pipe
            v8f acc0 = {}, acc1 = {};
            auto mm = [&](int s, v8f acc) -> v8f {
                // 16-bit B 32x16, contiguous k order used here (A matches): N = lane&15; half h -> K = 32*s + 16*(lane>>4) + h
                v16h bhv, blv;
                int base = (ct * 16 + lr) * D + 32 * s + g * 16;   // halves, 16-aligned
                ((uint4*)&bhv)[0] = Bh4[base >> 3];
                ((uint4*)&bhv)[1] = Bh4[(base >> 3) + 1];
                ((uint4*)&blv)[0] = Bl4[base >> 3];
                ((uint4*)&blv)[1] = Bl4[(base >> 3) + 1];
                acc = __builtin_amdgcn_wmma_f32_16x16x32_f16(false, aHi[s], false, bhv, (short)0, acc, false, false);
                acc = __builtin_amdgcn_wmma_f32_16x16x32_f16(false, aHi[s], false, blv, (short)0, acc, false, false);
                acc = __builtin_amdgcn_wmma_f32_16x16x32_f16(false, aLo[s], false, bhv, (short)0, acc, false, false);
                return acc;
            };
            acc0 = mm(0, acc0);
            acc1 = mm(1, acc1);
            acc0 = mm(2, acc0);
            acc1 = mm(3, acc1);

            // score = ||c||^2 - 2 x.c ; C/D layout: M = r + 8*(lane>>4), N = lane&15
            int   code = chunk * KT + ct * 16 + lr;
            float cnv  = cnorm[code];                 // L2-resident scalar per lane
#pragma unroll
            for (int r = 0; r < 8; ++r) {
                float sc = cnv - 2.0f * (acc0[r] + acc1[r]);
                if (sc < minv[r]) { minv[r] = sc; mini[r] = code; }  // strict <: first-index tie-break
            }
        }
        __syncthreads();            // everyone done with this buffer before chunk+2 overwrites it
    }

    // ---- argmin reduction across the 16 lanes of each half-wave group ----
#pragma unroll
    for (int m = 8; m >= 1; m >>= 1) {
#pragma unroll
        for (int r = 0; r < 8; ++r) {
            float ov = __shfl_xor(minv[r], m, 32);
            int   oi = __shfl_xor(mini[r], m, 32);
            if (ov < minv[r] || (ov == minv[r] && oi < mini[r])) { minv[r] = ov; mini[r] = oi; }
        }
    }

    // ---- gather winning codebook rows (f32, from L2) and write output ----
#pragma unroll
    for (int r = 0; r < 8; ++r) {
        long row = rowBase + wave * 16 + g * 8 + r;
        const float* q = cb + (long)mini[r] * D;
        float* o = out + row * D;
#pragma unroll
        for (int t = 0; t < 8; ++t) o[lr + 16 * t] = q[lr + 16 * t];
    }
}

extern "C" void kernel_launch(void* const* d_in, const int* in_sizes, int n_in,
                              void* d_out, int out_size, void* d_ws, size_t ws_size,
                              hipStream_t stream) {
    (void)n_in; (void)out_size; (void)ws_size;
    const float* enc = (const float*)d_in[0];   // [294912*128] f32
    const float* cb  = (const float*)d_in[1];   // [512*128] f32
    float* out = (float*)d_out;                 // same shape as enc, f32

    // workspace: cbHi (128 KB) | cbLo (128 KB) | cnorm (2 KB)
    _Float16* cbHi  = (_Float16*)d_ws;
    _Float16* cbLo  = cbHi + (size_t)KCODES * D;
    float*    cnorm = (float*)(cbLo + (size_t)KCODES * D);

    vq_prep_kernel<<<KCODES, 128, 0, stream>>>(cb, cbHi, cbLo, cnorm);

    const int nrows = in_sizes[0] / D;          // 294912, divisible by MT
    const int grid  = nrows / MT;               // 2304 workgroups
    const size_t lds = (size_t)(2 * MT * D + 2 * 2 * KT * D) * sizeof(_Float16); // 192 KB
    vq_main_kernel<<<grid, 256, lds, stream>>>(enc, cb, cbHi, cbLo, cnorm, out);
}
